// RNN_72353019068566
// MI455X (gfx1250) — matrix operand order repaired and checked by reading, not compiled, not run on hardware
//
#include <hip/hip_runtime.h>
#include <hip/hip_bf16.h>
#include <math.h>

// ---------------------------------------------------------------------------
// Problem constants: EMB=512 HID=1024 SEQ=128 BATCH=64 VOCAB=10000 LAYERS=2
// ---------------------------------------------------------------------------
static constexpr int kEmb   = 512;
static constexpr int kHid   = 1024;
static constexpr int kSeq   = 128;
static constexpr int kBatch = 64;
static constexpr int kVocab = 10000;   // 625 * 16 -> exact WMMA tiling

typedef __bf16 bf16_t;
typedef bf16_t bf16x8 __attribute__((ext_vector_type(8)));
typedef bf16_t v16bf  __attribute__((ext_vector_type(16)));
typedef float  v8f    __attribute__((ext_vector_type(8)));
typedef unsigned int u32;
typedef u32 v4u __attribute__((ext_vector_type(4)));
typedef int  v4i __attribute__((ext_vector_type(4)));
typedef int  v8i __attribute__((ext_vector_type(8)));

union FragU { v16bf v; bf16x8 h[2]; };

// A fragment 16x32 bf16 (MxK): lane r=lane&15 is the M row, half=lane>>4.
// lanes 0-15 hold K 0..7 / 16..23; lanes 16-31 hold K 8..15 / 24..31
// => two 16B runs at kb + 8*half and kb + 16 + 8*half.
__device__ __forceinline__ v16bf load_a_frag(const bf16_t* __restrict__ row, int kb, int half) {
  FragU u;
  u.h[0] = *reinterpret_cast<const bf16x8*>(row + kb + 8 * half);
  u.h[1] = *reinterpret_cast<const bf16x8*>(row + kb + 16 + 8 * half);
  return u.v;
}

// B fragment 32x16 bf16 (KxN): lane c=lane&15 is the N column; lanes 0-15 hold
// K 0..7 / 16..23, lanes 16-31 hold K 8..15 / 24..31 => two 16B runs at kb + 8*half and kb + 16 + 8*half.
// Weight is W[N,K] row-major, so lane c reads contiguous K from row (n0+c).
__device__ __forceinline__ v16bf load_b_frag(const bf16_t* __restrict__ col, int kb, int half) {
  FragU u;
  u.h[0] = *reinterpret_cast<const bf16x8*>(col + kb + 8 * half);
  u.h[1] = *reinterpret_cast<const bf16x8*>(col + kb + 16 + 8 * half);
  return u.v;
}

// ---------------------------------------------------------------------------
// Tensor Data Mover: 2-D tile load global -> LDS.
// D# bit packing per CDNA5 ISA §8.3 (group0) / §8.4 (group1):
//   g0.w0 = count=1; g0.w1 = lds_addr; g0.w2/w3 = global_addr[56:0]; type=2.
//   g1.w0 = data_size=2B (bits17:16=1) | pad_enable<<20 | pad_interval<<22
//           | pad_amount<<25 ; workgroup_mask[15:0]=0 (not in a cluster).
//   tensor_dim0 @bits79:48, tensor_dim1 @111:80, tile_dim0 @127:112,
//   tile_dim1 @143:128, tile_dim2 @159:144=0, dim0_stride @207:160.
// pad: row = tile_d0(=32 bf16) = 16 DWORDs (pad_interval=3), pad 4 DWORDs
// (pad_amount=3) -> LDS row stride of 40 bf16, matching the fragment readers.
// ---------------------------------------------------------------------------
__device__ __forceinline__ void tdm_load_2d(u32 lds_off, unsigned long long gaddr,
                                            u32 tensor_d0, u32 tensor_d1,
                                            u32 tile_d0, u32 tile_d1,
                                            u32 stride0_elems,
                                            u32 pad_interval, u32 pad_amount) {
  v4u g0;
  g0[0] = 1u;                                            // count=1, user mode
  g0[1] = lds_off;                                       // lds_addr (bytes)
  g0[2] = (u32)(gaddr & 0xffffffffull);                  // global_addr[31:0]
  g0[3] = (u32)((gaddr >> 32) & 0x01ffffffull) | (2u << 30); // addr[56:32], type=2
  v8i g1;
  u32 w0 = (1u << 16);                                   // data_size = 2 bytes
  w0 |= (1u << 20) | (pad_interval << 22) | (pad_amount << 25); // LDS padding
  g1[0] = (int)w0;
  g1[1] = (int)((tensor_d0 & 0xffffu) << 16);            // dim0[15:0] @ bit48
  g1[2] = (int)(((tensor_d0 >> 16) & 0xffffu) | ((tensor_d1 & 0xffffu) << 16));
  g1[3] = (int)(((tensor_d1 >> 16) & 0xffffu) | ((tile_d0 & 0xffffu) << 16));
  g1[4] = (int)(tile_d1 & 0xffffu);                      // tile_dim1; tile_dim2=0
  g1[5] = (int)stride0_elems;                            // dim0_stride[31:0]
  g1[6] = 0;                                             // stride0[47:32], stride1 lo
  g1[7] = 0;                                             // stride1 hi
  v4i g2 = {0, 0, 0, 0};                                 // unused (2-D tensor)
  v4i g3 = {0, 0, 0, 0};
#if __has_builtin(__builtin_amdgcn_tensor_load_to_lds)
#if defined(__clang_major__) && (__clang_major__ >= 23)
  v8i g4 = {0, 0, 0, 0, 0, 0, 0, 0};
  __builtin_amdgcn_tensor_load_to_lds(g0, g1, g2, g3, g4, 0);
#else
  __builtin_amdgcn_tensor_load_to_lds(g0, g1, g2, g3, 0);
#endif
#endif
}

// ---------------------------------------------------------------------------
// Prep kernels
// ---------------------------------------------------------------------------
__global__ void f32_to_bf16_kernel(bf16_t* __restrict__ dst, const float* __restrict__ src, int n) {
  int stride = gridDim.x * blockDim.x;
  for (int i = blockIdx.x * blockDim.x + threadIdx.x; i < n; i += stride)
    dst[i] = (bf16_t)src[i];
}

__global__ void gather_emb_kernel(bf16_t* __restrict__ X, const float* __restrict__ emb,
                                  const int* __restrict__ toks, int n) {
  int stride = gridDim.x * blockDim.x;
  for (int i = blockIdx.x * blockDim.x + threadIdx.x; i < n; i += stride) {
    int row = i >> 9;          // / kEmb
    int e   = i & (kEmb - 1);
    X[i] = (bf16_t)emb[(size_t)toks[row] * kEmb + e];
  }
}

__global__ void final_hidden_kernel(float* __restrict__ dst,
                                    const bf16_t* __restrict__ h0,
                                    const bf16_t* __restrict__ h1) {
  int i = blockIdx.x * blockDim.x + threadIdx.x;
  if (i < kBatch * kHid)          dst[i] = (float)h0[i];
  else if (i < 2 * kBatch * kHid) dst[i] = (float)h1[i - kBatch * kHid];
}

// ---------------------------------------------------------------------------
// Recurrent step GEMM:  dst = tanh( [A1 | A2] @ W^T + bias ),  M=64, N=1024.
// Branch-free: two K phases. One 16x16 tile per wave (256 waves) to maximize
// parallelism on the latency-bound sequential chain. (Codegen verified: clause'd
// b128 loads with partial loadcnt waits overlapping WMMAs.)
// ---------------------------------------------------------------------------
__global__ void __launch_bounds__(256) rnn_step_kernel(
    const bf16_t* __restrict__ A1, int K1,
    const bf16_t* __restrict__ A2, int K2,
    const bf16_t* __restrict__ W, const float* __restrict__ bias,
    bf16_t* __restrict__ dst0, bf16_t* __restrict__ dst1) {
  const int lane = threadIdx.x & 31;
  const int half = lane >> 4;
  const int r    = lane & 15;
  const int gid  = blockIdx.x * 8 + (threadIdx.x >> 5);   // 0..255
  const int mt   = gid & 3;                               // 4 M tiles (M=64)
  const int n0   = (gid >> 2) * 16;                       // 64 N tiles (N=1024)
  const int K    = K1 + K2;

  const bf16_t* a1row = A1 + (size_t)(mt * 16 + r) * K1;
  const bf16_t* a2row = A2 + (size_t)(mt * 16 + r) * K2;
  const bf16_t* wrow  = W  + (size_t)(n0 + r) * K;

  v8f acc = {};
#pragma unroll 4
  for (int kb = 0; kb < K1; kb += 32) {
    v16bf a = load_a_frag(a1row, kb, half);
    v16bf b = load_b_frag(wrow, kb, half);
    acc = __builtin_amdgcn_wmma_f32_16x16x32_bf16(false, a, false, b, (short)0, acc, false, false);
  }
#pragma unroll 4
  for (int kb = 0; kb < K2; kb += 32) {
    v16bf a = load_a_frag(a2row, kb, half);
    v16bf b = load_b_frag(wrow, K1 + kb, half);
    acc = __builtin_amdgcn_wmma_f32_16x16x32_bf16(false, a, false, b, (short)0, acc, false, false);
  }

  // C/D layout: element (m = v + 8*half, n = lane&15) lives in acc[v].
  const int col = n0 + r;
  const float bv = bias[col];
#pragma unroll
  for (int v = 0; v < 8; ++v) {
    int row = mt * 16 + v + 8 * half;
    bf16_t o = (bf16_t)tanhf(acc[v] + bv);
    dst0[(size_t)row * kHid + col] = o;
    if (dst1) dst1[(size_t)row * kHid + col] = o;
  }
}

// ---------------------------------------------------------------------------
// Logits GEMM: out[8192,10000] = H[8192,1024] @ Wl[10000,1024]^T + bl.
// Block tile 128(M) x 80(N), BK=32, 8 waves (wave w -> M rows 16w..16w+15,
// all 80 N). TDM double-buffering: wave 0 issues tensor_load_to_lds for the
// next K-tile (As: 128x32, Bs: 80x32, both padded to 40-elem LDS stride ->
// conflict-free b128 fragment reads), waves compute 5 WMMAs per tile from LDS,
// then wave 0 waits TENSORcnt and the block barriers.
// Grid: (512/8) M-blocks x 125 N-blocks = 8000 blocks x 256 threads.
// ---------------------------------------------------------------------------
__global__ void __launch_bounds__(256) logits_kernel(
    const bf16_t* __restrict__ H, const bf16_t* __restrict__ Wl,
    const float* __restrict__ bl, float* __restrict__ out) {
  constexpr int LDA = 40;                 // padded K-stride (elements)
  __shared__ bf16_t As[2][128 * LDA];     // 20.0 KB
  __shared__ bf16_t Bs[2][80 * LDA];      // 12.5 KB

  const int tid  = threadIdx.x;
  const int lane = tid & 31;
  const int half = lane >> 4;
  const int r    = lane & 15;
  const int w    = tid >> 5;

  const int m0 = (blockIdx.x / 125) * 128;
  const int n0 = (blockIdx.x % 125) * 80;

  // Uniform (SGPR) tile-origin addresses and LDS offsets for the TDM.
  const unsigned long long gA = (unsigned long long)(H  + (size_t)m0 * kHid);
  const unsigned long long gB = (unsigned long long)(Wl + (size_t)n0 * kHid);
  const u32 ldsA[2] = {(u32)(size_t)&As[0][0], (u32)(size_t)&As[1][0]};
  const u32 ldsB[2] = {(u32)(size_t)&Bs[0][0], (u32)(size_t)&Bs[1][0]};

  // Prologue: wave 0 DMAs K-tile 0 into buffer 0 and waits for it.
  if (tid < 32) {
    tdm_load_2d(ldsA[0], gA, kHid, 128, 32, 128, kHid, 3, 3);
    tdm_load_2d(ldsB[0], gB, kHid, 80,  32, 80,  kHid, 3, 3);
    __builtin_amdgcn_s_wait_tensorcnt(0);
  }
  __syncthreads();

  v8f acc[5] = {};
  constexpr int KT = kHid / 32;           // 32 K-tiles
#pragma unroll 1
  for (int kt = 0; kt < KT; ++kt) {
    const int cur = kt & 1, nxt = cur ^ 1;
    const bool has_next = (kt + 1) < KT;

    // Kick off the next tile's DMA; it lands in the other buffer while the
    // whole block computes on the current one.
    if (has_next && tid < 32) {
      const unsigned long long koff = (unsigned long long)(kt + 1) * 32ull * 2ull;
      tdm_load_2d(ldsA[nxt], gA + koff, kHid, 128, 32, 128, kHid, 3, 3);
      tdm_load_2d(ldsB[nxt], gB + koff, kHid, 80,  32, 80,  kHid, 3, 3);
    }

    // Compute from the current LDS buffer.
    FragU a;
    const bf16_t* as = &As[cur][(w * 16 + r) * LDA];
    a.h[0] = *(const bf16x8*)(as + 8 * half);
    a.h[1] = *(const bf16x8*)(as + 16 + 8 * half);
#pragma unroll
    for (int j = 0; j < 5; ++j) {
      FragU b;
      const bf16_t* bs = &Bs[cur][(j * 16 + r) * LDA];
      b.h[0] = *(const bf16x8*)(bs + 8 * half);
      b.h[1] = *(const bf16x8*)(bs + 16 + 8 * half);
      acc[j] = __builtin_amdgcn_wmma_f32_16x16x32_bf16(false, a.v, false, b.v,
                                                       (short)0, acc[j], false, false);
    }

    if (has_next) {
      if (tid < 32) __builtin_amdgcn_s_wait_tensorcnt(0);
      __syncthreads();
    }
  }

  // Epilogue: bias add, fp32 store.
#pragma unroll
  for (int j = 0; j < 5; ++j) {
    const int col = n0 + j * 16 + r;
    const float bv = bl[col];
#pragma unroll
    for (int v = 0; v < 8; ++v) {
      const int row = m0 + w * 16 + v + 8 * half;
      out[(size_t)row * kVocab + col] = acc[j][v] + bv;
    }
  }
}

// ---------------------------------------------------------------------------
// Host orchestration
// ---------------------------------------------------------------------------
extern "C" void kernel_launch(void* const* d_in, const int* in_sizes, int n_in,
                              void* d_out, int out_size, void* d_ws, size_t ws_size,
                              hipStream_t stream) {
  (void)in_sizes; (void)n_in; (void)out_size; (void)ws_size;
  const int*   inputs = (const int*)  d_in[0];   // [T, B]
  const float* hidden = (const float*)d_in[1];   // [2, B, H]
  const float* emb    = (const float*)d_in[2];   // [V, E]
  const float* W0     = (const float*)d_in[3];   // [H, E+H]
  const float* b0     = (const float*)d_in[4];   // [H]
  const float* Wm     = (const float*)d_in[5];   // [H, 2H]
  const float* bm     = (const float*)d_in[6];   // [H]
  const float* Wl     = (const float*)d_in[7];   // [V, H]
  const float* bl     = (const float*)d_in[8];   // [V]
  float* out = (float*)d_out;                    // logits [T,B,V] then h_final [2,B,H]

  char* ws = (char*)d_ws;
  size_t off = 0;
  auto carve = [&](size_t bytes) -> char* {
    char* p = ws + off;
    off = (off + bytes + 255) & ~(size_t)255;
    return p;
  };
  bf16_t* W0b   = (bf16_t*)carve((size_t)kHid * (kEmb + kHid) * 2);  // 3.1 MB
  bf16_t* Wmb   = (bf16_t*)carve((size_t)kHid * 2 * kHid * 2);       // 4.2 MB
  bf16_t* Wlb   = (bf16_t*)carve((size_t)kVocab * kHid * 2);         // 20.5 MB
  bf16_t* Xemb  = (bf16_t*)carve((size_t)kSeq * kBatch * kEmb * 2);  // 8.4 MB
  bf16_t* H1all = (bf16_t*)carve((size_t)kSeq * kBatch * kHid * 2);  // 16.8 MB
  bf16_t* h0buf[2] = {(bf16_t*)carve(kBatch * kHid * 2), (bf16_t*)carve(kBatch * kHid * 2)};
  bf16_t* h1buf[2] = {(bf16_t*)carve(kBatch * kHid * 2), (bf16_t*)carve(kBatch * kHid * 2)};

  // --- prep: weight/activation conversion + embedding gather ---
  f32_to_bf16_kernel<<<2048, 256, 0, stream>>>(W0b, W0, kHid * (kEmb + kHid));
  f32_to_bf16_kernel<<<2048, 256, 0, stream>>>(Wmb, Wm, kHid * 2 * kHid);
  f32_to_bf16_kernel<<<4096, 256, 0, stream>>>(Wlb, Wl, kVocab * kHid);
  f32_to_bf16_kernel<<<256,  256, 0, stream>>>(h0buf[0], hidden, kBatch * kHid);
  f32_to_bf16_kernel<<<256,  256, 0, stream>>>(h1buf[0], hidden + kBatch * kHid, kBatch * kHid);
  gather_emb_kernel<<<4096, 256, 0, stream>>>(Xemb, emb, inputs, kSeq * kBatch * kEmb);

  // --- sequential recurrence (ping-pong hidden buffers; initial state in [0]) ---
  for (int t = 0; t < kSeq; ++t) {
    int p = t & 1, q = p ^ 1;
    // layer 0: h0' = tanh([x_t | h0] W0^T + b0)
    rnn_step_kernel<<<32, 256, 0, stream>>>(
        Xemb + (size_t)t * kBatch * kEmb, kEmb,
        h0buf[p], kHid, W0b, b0, h0buf[q], (bf16_t*)nullptr);
    // layer 1: h1' = tanh([h0' | h1] Wm^T + bm), stash into H1all[t]
    rnn_step_kernel<<<32, 256, 0, stream>>>(
        h0buf[q], kHid,
        h1buf[p], kHid, Wmb, bm, h1buf[q], H1all + (size_t)t * kBatch * kHid);
  }

  // --- one big logits GEMM over all timesteps: [8192,1024] x [10000,1024]^T ---
  logits_kernel<<<8000, 256, 0, stream>>>(H1all, Wlb, bl, out);

  // --- final hidden state (kSeq even => final state sits in buffer index 0) ---
  final_hidden_kernel<<<(2 * kBatch * kHid + 255) / 256, 256, 0, stream>>>(
      out + (size_t)kSeq * kBatch * kVocab, h0buf[0], h1buf[0]);
}
